// HunYuanSparseMoeBlock_34376918237697
// MI455X (gfx1250) — compile-verified, matrix operand order checked by reading, not run on hardware
//
#include <hip/hip_runtime.h>
#include <hip/hip_bf16.h>

// ---- problem constants (from reference) ----
#define T_TOK 4096   // B*S tokens
#define H_DIM 2048
#define I_DIM 4096
#define E_NUM 8
#define K_TOP 2

typedef __bf16 bf16_t;
typedef __attribute__((ext_vector_type(16))) __bf16 v16bf;
typedef __attribute__((ext_vector_type(8)))  __bf16 v8bf;
typedef __attribute__((ext_vector_type(8)))  float  v8f;

// TDM descriptor SGPR-group types (probe-confirmed builtin arity)
typedef __attribute__((ext_vector_type(4))) unsigned int tdm_g0_t;
typedef __attribute__((ext_vector_type(8))) int          tdm_g1_t;
typedef __attribute__((ext_vector_type(4))) int          tdm_g2_t;

// ---- tiling ----
constexpr int BM  = 128;  // rows (token slots) per block
constexpr int BN  = 64;   // cols per block (per tensor)
constexpr int BK  = 32;   // K slab == WMMA K for bf16
constexpr int LDT = 40;   // bf16 LDS row stride (32 + 8 pad -> 80B, 16B aligned)
constexpr int LDF = 65;   // f32 staging row stride (64 + 1 pad dword)

// Fragment loader following CDNA5 16-bit matrix layout (bf16 LDS, b128 loads):
// lanes 0-15 : row = row0+lane,    K chunks 0..7 and 16..23
// lanes 16-31: row = row0+lane-16, K chunks 8..15 and 24..31
__device__ inline v16bf load_frag(const bf16_t* base, int row0, int lane) {
    const int r  = row0 + (lane & 15);
    const int kc = (lane >> 4) << 3;          // 0 or 8
    const bf16_t* p = base + r * LDT + kc;    // 16B-aligned by construction
    v8bf lo = *(const v8bf*)(p);
    v8bf hi = *(const v8bf*)(p + 16);
    return __builtin_shufflevector(lo, hi, 0,1,2,3,4,5,6,7,8,9,10,11,12,13,14,15);
}

// B fragment built from the TDM-staged f32 tile (row-major [32][LDF]),
// converting f32 -> bf16 per element. Lane holds column n0 + (lane&15).
__device__ inline v16bf load_frag_f32cvt(const float* base, int n0, int lane) {
    const int n  = n0 + (lane & 15);
    const int kc = (lane >> 4) << 3;          // 0 or 8
    v16bf f;
#pragma unroll
    for (int i = 0; i < 8; ++i) {
        f[i]     = (bf16_t)base[(kc + i)      * LDF + n];
        f[8 + i] = (bf16_t)base[(kc + 16 + i) * LDF + n];
    }
    return f;
}

__device__ inline v8f wmma_bf16(v16bf a, v16bf b, v8f c) {
    // (neg_a, A, neg_b, B, c_mod, C, reuse_a, reuse_b)
    return __builtin_amdgcn_wmma_f32_16x16x32_bf16(false, a, false, b, (short)0, c, false, false);
}

__device__ inline unsigned lds_off(const void* p) {
    return (unsigned)(uintptr_t)p;   // LDS flat addr: upper bits discarded by HW
}

// Issue a 2-D TENSOR_LOAD_TO_LDS. Descriptor per ISA cdna5 ch.8:
//  g0: [1:0]=count=1 | [63:32]=lds_addr | [120:64]=global_addr | [127:126]=type(2)
//  g1: [17:16]=data_size | [20]=pad_en | [24:22]=pad_interval | [31:25]=pad_amount
//      [79:48]=tensor_dim0 | [111:80]=tensor_dim1 | [127:112]=tile_dim0
//      [143:128]=tile_dim1 | [207:160]=tensor_dim0_stride
__device__ inline void tdm_load_2d(unsigned lds_addr, const void* gaddr,
                                   unsigned ds_code,          // 0=1B 1=2B 2=4B
                                   unsigned tile_x, unsigned tile_y,
                                   unsigned tensor_x, unsigned tensor_y,
                                   unsigned long long stride_x,
                                   unsigned pad_en, unsigned pad_int, unsigned pad_amt) {
    const unsigned long long ga = (unsigned long long)(uintptr_t)gaddr;
    tdm_g0_t g0;
    g0.x = 1u;                                                 // count=1, user descriptor
    g0.y = lds_addr;
    g0.z = (unsigned)(ga & 0xffffffffu);
    g0.w = (unsigned)((ga >> 32) & 0x1ffffffu) | (2u << 30);   // addr[56:32] | type=2
    tdm_g1_t g1;
    g1[0] = (int)((ds_code << 16) | (pad_en << 20) | (pad_int << 22) | (pad_amt << 25));
    g1[1] = (int)((tensor_x & 0xffffu) << 16);                 // tensor_dim0[15:0]
    g1[2] = (int)(((tensor_x >> 16) & 0xffffu) | ((tensor_y & 0xffffu) << 16));
    g1[3] = (int)(((tensor_y >> 16) & 0xffffu) | ((tile_x & 0xffffu) << 16));
    g1[4] = (int)(tile_y & 0xffffu);                           // tile_dim1 (tile_dim2=0)
    g1[5] = (int)(unsigned)(stride_x & 0xffffffffull);
    g1[6] = (int)(unsigned)((stride_x >> 32) & 0xffffull);     // stride hi16 (dim1_stride=0)
    g1[7] = 0;
    const tdm_g2_t gz = {0, 0, 0, 0};
#if defined(__clang_major__) && (__clang_major__ >= 23)
    const tdm_g1_t gz8 = {0, 0, 0, 0, 0, 0, 0, 0};
    __builtin_amdgcn_tensor_load_to_lds(g0, g1, gz, gz, gz8, 0);
#else
    __builtin_amdgcn_tensor_load_to_lds(g0, g1, gz, gz, 0);
#endif
}

// ---------------------------------------------------------------------------
// 0) init: zero output, cast x -> bf16, zero per-expert counters
// ---------------------------------------------------------------------------
__global__ __launch_bounds__(256) void init_kernel(const float* __restrict__ x,
                                                   float* __restrict__ out,
                                                   bf16_t* __restrict__ xb,
                                                   int* __restrict__ counts,
                                                   size_t n) {
    size_t i = (size_t)blockIdx.x * 256 + threadIdx.x;
    if (i < n) {
        out[i] = 0.0f;
        xb[i]  = (bf16_t)x[i];
    }
    if (i < E_NUM) counts[i] = 0;
}

// ---------------------------------------------------------------------------
// 1) router: logits = x @ gate_w, softmax, top-2, renormalize
// ---------------------------------------------------------------------------
__global__ __launch_bounds__(256) void router_kernel(const float* __restrict__ x,
                                                     const float* __restrict__ gate_w,
                                                     int* __restrict__ counts,
                                                     int* __restrict__ topk_e,
                                                     float* __restrict__ topk_w) {
    const int t   = blockIdx.x;
    const int tid = threadIdx.x;
    float part[E_NUM];
#pragma unroll
    for (int e = 0; e < E_NUM; ++e) part[e] = 0.0f;

    const float* xr = x + (size_t)t * H_DIM;
    for (int h = tid; h < H_DIM; h += 256) {
        const float xv = xr[h];
        const float* gr = gate_w + (size_t)h * E_NUM;
#pragma unroll
        for (int e = 0; e < E_NUM; ++e) part[e] += xv * gr[e];
    }

    __shared__ float red[E_NUM][256];
#pragma unroll
    for (int e = 0; e < E_NUM; ++e) red[e][tid] = part[e];
    __syncthreads();
    for (int s = 128; s > 0; s >>= 1) {
        if (tid < s) {
#pragma unroll
            for (int e = 0; e < E_NUM; ++e) red[e][tid] += red[e][tid + s];
        }
        __syncthreads();
    }

    if (tid == 0) {
        float lg[E_NUM], p[E_NUM];
        float mx = -1e30f;
#pragma unroll
        for (int e = 0; e < E_NUM; ++e) { lg[e] = red[e][0]; mx = fmaxf(mx, lg[e]); }
#pragma unroll
        for (int e = 0; e < E_NUM; ++e) p[e] = __expf(lg[e] - mx);
        int i1 = 0;
#pragma unroll
        for (int e = 1; e < E_NUM; ++e) if (p[e] > p[i1]) i1 = e;
        int i2 = (i1 == 0) ? 1 : 0;
#pragma unroll
        for (int e = 0; e < E_NUM; ++e) if (e != i1 && p[e] > p[i2]) i2 = e;
        const float s2 = p[i1] + p[i2];
        topk_e[t * 2 + 0] = i1;
        topk_e[t * 2 + 1] = i2;
        topk_w[t * 2 + 0] = p[i1] / s2;   // softmax denom cancels in renorm
        topk_w[t * 2 + 1] = p[i2] / s2;
        atomicAdd(&counts[i1], 1);
        atomicAdd(&counts[i2], 1);
    }
}

// ---------------------------------------------------------------------------
// 2) prefix over expert counts (tiny)
// ---------------------------------------------------------------------------
__global__ void prefix_kernel(const int* __restrict__ counts,
                              int* __restrict__ cursor,
                              int* __restrict__ prefix) {
    if (threadIdx.x == 0 && blockIdx.x == 0) {
        int acc = 0;
        for (int e = 0; e < E_NUM; ++e) {
            prefix[e] = acc;
            acc += counts[e];
            cursor[e] = 0;
        }
        prefix[E_NUM] = acc;   // == K_TOP * T_TOK
    }
}

// ---------------------------------------------------------------------------
// 3) dispatch: build per-expert compacted (token, weight) lists; shared = all
// ---------------------------------------------------------------------------
__global__ __launch_bounds__(256) void dispatch_kernel(const int* __restrict__ topk_e,
                                                       const float* __restrict__ topk_w,
                                                       const int* __restrict__ prefix,
                                                       int* __restrict__ cursor,
                                                       int* __restrict__ tok,
                                                       float* __restrict__ twt) {
    const int t = blockIdx.x * 256 + threadIdx.x;
    if (t >= T_TOK) return;
#pragma unroll
    for (int k = 0; k < K_TOP; ++k) {
        const int   e = topk_e[t * 2 + k];
        const float w = topk_w[t * 2 + k];
        const int pos = atomicAdd(&cursor[e], 1);
        const int idx = prefix[e] + pos;
        tok[idx] = t;
        twt[idx] = w;
    }
    tok[K_TOP * T_TOK + t] = t;     // shared-expert region
    twt[K_TOP * T_TOK + t] = 1.0f;
}

// ---------------------------------------------------------------------------
// 4) GEMM1: act = silu(Xg @ Wg) * (Xg @ Wu), bf16 out.  z-dim = expert (8=shared)
//    A: gathered bf16 rows (manual). B: double-buffered pipelined TDM tiles.
// ---------------------------------------------------------------------------
__global__ __launch_bounds__(256) void gemm1_kernel(const bf16_t* __restrict__ xb,
                                                    const float* __restrict__ w_gate_up,
                                                    const float* __restrict__ shared_gate_up,
                                                    const int* __restrict__ counts,
                                                    const int* __restrict__ prefix,
                                                    const int* __restrict__ tok,
                                                    bf16_t* __restrict__ act) {
    const int e   = blockIdx.z;
    const int cnt = (e == E_NUM) ? T_TOK : counts[e];
    const int by  = blockIdx.y;
    if (by * BM >= cnt) return;
    const int base = (e == E_NUM) ? (K_TOP * T_TOK) : prefix[e];
    const float* W = (e == E_NUM) ? shared_gate_up
                                  : (w_gate_up + (size_t)e * H_DIM * (2 * I_DIM));
    const int col0 = blockIdx.x * BN;

    __shared__ __align__(16) bf16_t As[BM * LDT];
    __shared__ __align__(16) float  BgF[2][BK * LDF];   // double-buffered TDM staging (g)
    __shared__ __align__(16) float  BuF[2][BK * LDF];   // double-buffered TDM staging (u)

    const int tid  = threadIdx.x;
    const int lane = tid & 31;
    const int wave = tid >> 5;
    const int wm   = (wave & 3) * 32;
    const int wn   = (wave >> 2) * 32;

    // A staging: 2 threads per row, 16 bf16 each (gathered token rows)
    const int ar = tid >> 1, ah = (tid & 1) * 16;
    const int aslot = by * BM + ar;
    const int atokn = (aslot < cnt) ? tok[base + aslot] : 0;
    const bf16_t* arow = xb + (size_t)atokn * H_DIM;

    v8f cg[2][2] = {};
    v8f cu[2][2] = {};

    // TDM pipeline prologue: stage 0 tiles in flight before the loop
    if (wave == 0) {
        tdm_load_2d(lds_off(BgF[0]), W + col0,
                    /*ds=*/2, BN, BK, 2 * I_DIM, H_DIM, 2 * I_DIM, 1, /*64dw*/5, /*1dw*/0);
        tdm_load_2d(lds_off(BuF[0]), W + I_DIM + col0,
                    2, BN, BK, 2 * I_DIM, H_DIM, 2 * I_DIM, 1, 5, 0);
    }

    int cur = 0;
    for (int kb = 0; kb < H_DIM; kb += BK, cur ^= 1) {
        // A tile (bf16, manual) -- safe: previous stage ended with a barrier
        *(v8bf*)&As[ar * LDT + ah]     = *(const v8bf*)(arow + kb + ah);
        *(v8bf*)&As[ar * LDT + ah + 8] = *(const v8bf*)(arow + kb + ah + 8);

        // issue next stage's DMAs, then drain current stage (in-order TENSORcnt)
        const int kn = kb + BK;
        if (wave == 0) {
            if (kn < H_DIM) {
                tdm_load_2d(lds_off(BgF[cur ^ 1]), W + (size_t)kn * (2 * I_DIM) + col0,
                            2, BN, BK, 2 * I_DIM, H_DIM, 2 * I_DIM, 1, 5, 0);
                tdm_load_2d(lds_off(BuF[cur ^ 1]), W + (size_t)kn * (2 * I_DIM) + I_DIM + col0,
                            2, BN, BK, 2 * I_DIM, H_DIM, 2 * I_DIM, 1, 5, 0);
                __builtin_amdgcn_s_wait_tensorcnt(2);   // next 2 in flight, cur 2 landed
            } else {
                __builtin_amdgcn_s_wait_tensorcnt(0);
            }
        }
        if (kn < H_DIM)
            __builtin_prefetch(W + (size_t)kn * (2 * I_DIM) + col0 + lane, 0, 1);
        __syncthreads();

        v16bf a0 = load_frag(As, wm,      lane);
        v16bf a1 = load_frag(As, wm + 16, lane);
        v16bf g0 = load_frag_f32cvt(BgF[cur], wn,      lane);
        v16bf g1 = load_frag_f32cvt(BgF[cur], wn + 16, lane);
        v16bf u0 = load_frag_f32cvt(BuF[cur], wn,      lane);
        v16bf u1 = load_frag_f32cvt(BuF[cur], wn + 16, lane);

        cg[0][0] = wmma_bf16(a0, g0, cg[0][0]);
        cg[0][1] = wmma_bf16(a0, g1, cg[0][1]);
        cg[1][0] = wmma_bf16(a1, g0, cg[1][0]);
        cg[1][1] = wmma_bf16(a1, g1, cg[1][1]);
        cu[0][0] = wmma_bf16(a0, u0, cu[0][0]);
        cu[0][1] = wmma_bf16(a0, u1, cu[0][1]);
        cu[1][0] = wmma_bf16(a1, u0, cu[1][0]);
        cu[1][1] = wmma_bf16(a1, u1, cu[1][1]);
        __syncthreads();
    }

    // epilogue: silu(g)*u -> act (bf16), masked by row validity
    const int hb = (lane >> 4) * 8;
    const int cl = lane & 15;
#pragma unroll
    for (int mi = 0; mi < 2; ++mi)
#pragma unroll
        for (int ni = 0; ni < 2; ++ni)
#pragma unroll
            for (int i = 0; i < 8; ++i) {
                const int r  = wm + mi * 16 + hb + i;
                const int sr = by * BM + r;
                if (sr < cnt) {
                    const int col = col0 + wn + ni * 16 + cl;
                    const float g = cg[mi][ni][i];
                    const float u = cu[mi][ni][i];
                    const float s = g / (1.0f + __expf(-g));
                    act[(size_t)(base + sr) * I_DIM + col] = (bf16_t)(s * u);
                }
            }
}

// ---------------------------------------------------------------------------
// 5) GEMM2: out[token] += weight * (act @ Wd).  Launched per expert (serialized).
//    A (bf16) and B (f32) both double-buffered pipelined TDM tiles.
// ---------------------------------------------------------------------------
__global__ __launch_bounds__(256) void gemm2_kernel(const bf16_t* __restrict__ act,
                                                    const float* __restrict__ w_down,
                                                    const float* __restrict__ shared_down,
                                                    const int* __restrict__ counts,
                                                    const int* __restrict__ prefix,
                                                    const int* __restrict__ tok,
                                                    const float* __restrict__ twt,
                                                    float* __restrict__ out,
                                                    int e) {
    const int cnt = (e == E_NUM) ? T_TOK : counts[e];
    const int by  = blockIdx.y;
    if (by * BM >= cnt) return;
    const int base = (e == E_NUM) ? (K_TOP * T_TOK) : prefix[e];
    const float* W = (e == E_NUM) ? shared_down
                                  : (w_down + (size_t)e * I_DIM * H_DIM);
    const int col0 = blockIdx.x * BN;

    __shared__ __align__(16) bf16_t As[2][BM * LDT];    // TDM pads to LDT=40 stride
    __shared__ __align__(16) float  BsF[2][BK * LDF];   // TDM f32 staging

    const int tid  = threadIdx.x;
    const int lane = tid & 31;
    const int wave = tid >> 5;
    const int wm   = (wave & 3) * 32;
    const int wn   = (wave >> 2) * 32;

    const bf16_t* abase = act + (size_t)(base + by * BM) * I_DIM;

    v8f c[2][2] = {};

    if (wave == 0) {
        // A: 128 rows x 32 bf16, row stride I; pad 4 dwords per 16 -> LDT=40
        tdm_load_2d(lds_off(As[0]), abase,
                    /*ds=*/1, BK, BM, I_DIM, BM, I_DIM, 1, /*16dw*/3, /*4dw*/3);
        // B: 32 rows x 64 f32 from w_down, row stride H
        tdm_load_2d(lds_off(BsF[0]), W + col0,
                    2, BN, BK, H_DIM, I_DIM, H_DIM, 1, 5, 0);
    }

    int cur = 0;
    for (int kb = 0; kb < I_DIM; kb += BK, cur ^= 1) {
        const int kn = kb + BK;
        if (wave == 0) {
            if (kn < I_DIM) {
                tdm_load_2d(lds_off(As[cur ^ 1]), abase + kn,
                            1, BK, BM, I_DIM, BM, I_DIM, 1, 3, 3);
                tdm_load_2d(lds_off(BsF[cur ^ 1]), W + (size_t)kn * H_DIM + col0,
                            2, BN, BK, H_DIM, I_DIM, H_DIM, 1, 5, 0);
                __builtin_amdgcn_s_wait_tensorcnt(2);   // next 2 in flight, cur 2 landed
            } else {
                __builtin_amdgcn_s_wait_tensorcnt(0);
            }
        }
        if (kn < I_DIM)
            __builtin_prefetch(W + (size_t)kn * H_DIM + col0 + lane, 0, 1);
        __syncthreads();

        v16bf a0 = load_frag(As[cur], wm,      lane);
        v16bf a1 = load_frag(As[cur], wm + 16, lane);
        v16bf b0 = load_frag_f32cvt(BsF[cur], wn,      lane);
        v16bf b1 = load_frag_f32cvt(BsF[cur], wn + 16, lane);

        c[0][0] = wmma_bf16(a0, b0, c[0][0]);
        c[0][1] = wmma_bf16(a0, b1, c[0][1]);
        c[1][0] = wmma_bf16(a1, b0, c[1][0]);
        c[1][1] = wmma_bf16(a1, b1, c[1][1]);
        __syncthreads();
    }

    // epilogue: scatter-accumulate weight * value into output rows
    const int hb = (lane >> 4) * 8;
    const int cl = lane & 15;
#pragma unroll
    for (int mi = 0; mi < 2; ++mi)
#pragma unroll
        for (int ni = 0; ni < 2; ++ni)
#pragma unroll
            for (int i = 0; i < 8; ++i) {
                const int r  = wm + mi * 16 + hb + i;
                const int sr = by * BM + r;
                if (sr < cnt) {
                    const int   t   = tok[base + sr];
                    const float w   = twt[base + sr];
                    const int   col = col0 + wn + ni * 16 + cl;
                    out[(size_t)t * H_DIM + col] += w * c[mi][ni][i];
                }
            }
}

// ---------------------------------------------------------------------------
// launcher
// ---------------------------------------------------------------------------
extern "C" void kernel_launch(void* const* d_in, const int* in_sizes, int n_in,
                              void* d_out, int out_size, void* d_ws, size_t ws_size,
                              hipStream_t stream) {
    (void)in_sizes; (void)n_in; (void)out_size; (void)ws_size;

    const float* x   = (const float*)d_in[0];  // [T, H]
    const float* gw  = (const float*)d_in[1];  // [H, E]
    const float* wgu = (const float*)d_in[2];  // [E, H, 2I]
    const float* wd  = (const float*)d_in[3];  // [E, I, H]
    const float* sgu = (const float*)d_in[4];  // [H, 2I]
    const float* sd  = (const float*)d_in[5];  // [I, H]
    float* out = (float*)d_out;                // [T, H]

    // workspace carve-up (256B aligned regions)
    char* ws = (char*)d_ws;
    size_t off = 0;
    auto take = [&](size_t bytes) -> char* {
        char* p = ws + off;
        off += (bytes + 255) & ~(size_t)255;
        return p;
    };
    bf16_t* xb     = (bf16_t*)take((size_t)T_TOK * H_DIM * sizeof(bf16_t));
    bf16_t* act    = (bf16_t*)take((size_t)(K_TOP + 1) * T_TOK * I_DIM * sizeof(bf16_t));
    int*    counts = (int*)take(E_NUM * sizeof(int));
    int*    cursor = (int*)take(E_NUM * sizeof(int));
    int*    prefix = (int*)take((E_NUM + 1) * sizeof(int));
    int*    tok    = (int*)take((size_t)(K_TOP + 1) * T_TOK * sizeof(int));
    float*  twt    = (float*)take((size_t)(K_TOP + 1) * T_TOK * sizeof(float));
    int*    tke    = (int*)take((size_t)T_TOK * K_TOP * sizeof(int));
    float*  tkw    = (float*)take((size_t)T_TOK * K_TOP * sizeof(float));

    const size_t n = (size_t)T_TOK * H_DIM;
    init_kernel<<<dim3((unsigned)((n + 255) / 256)), 256, 0, stream>>>(x, out, xb, counts, n);
    router_kernel<<<dim3(T_TOK), 256, 0, stream>>>(x, gw, counts, tke, tkw);
    prefix_kernel<<<1, 32, 0, stream>>>(counts, cursor, prefix);
    dispatch_kernel<<<dim3(T_TOK / 256), 256, 0, stream>>>(tke, tkw, prefix, cursor, tok, twt);

    // GEMM1 over all 9 "experts" at once (disjoint act regions -> no races)
    gemm1_kernel<<<dim3(I_DIM / BN, T_TOK / BM, E_NUM + 1), 256, 0, stream>>>(
        xb, wgu, sgu, counts, prefix, tok, act);

    // GEMM2 per expert, stream-serialized so += into d_out is race-free and deterministic
    for (int e = 0; e <= E_NUM; ++e) {
        gemm2_kernel<<<dim3(H_DIM / BN, T_TOK / BM), 256, 0, stream>>>(
            act, wd, sd, counts, prefix, tok, twt, out, e);
    }
}
